// MultiHeadAttention_1769526526351
// MI455X (gfx1250) — matrix operand order repaired and checked by reading, not compiled, not run on hardware
//
#include <hip/hip_runtime.h>
#include <hip/hip_bf16.h>

// ---------------------------------------------------------------------------
// MHA forward for MI455X (gfx1250, wave32, WMMA bf16 16x16x32, f32 accum)
// B=2, S=2048, D=1024, H=16, HD=64.  d_out = [out (B*S*D) | attn (B*H*S*S)]
// Bandwidth-bound (~1.2 GB traffic); bf16 WMMA w/ f32 accum; attn written once.
// ---------------------------------------------------------------------------

typedef __bf16 bf16_t;
typedef __attribute__((ext_vector_type(16))) __bf16 v16bf;
typedef __attribute__((ext_vector_type(8)))  __bf16 v8bf;
typedef __attribute__((ext_vector_type(8)))  float  v8f;

static constexpr int B_  = 2;
static constexpr int S_  = 2048;
static constexpr int D_  = 1024;
static constexpr int H_  = 16;
static constexpr int HD_ = 64;
static constexpr float INV_SCALE = 0.125f; // 1/sqrt(64)

#define WMMA_BF16(a, b, c)                                                     \
  __builtin_amdgcn_wmma_f32_16x16x32_bf16(false, (a), false, (b), (short)0,    \
                                          (c), false, false)

// --- WMMA fragment loaders (layouts per CDNA5 ISA 7.12.2) --------
// A (16x32, MxK): lane = row (m = lane&15), k = g*16 + half*8 + j
__device__ __forceinline__ v16bf frag_a_from_f32(const float* tile, int ld) {
  const int lane = threadIdx.x & 31;
  const int m = lane & 15, hf = lane >> 4;
  const float* p = tile + (size_t)m * ld + hf * 8;
  v16bf a;
#pragma unroll
  for (int g = 0; g < 2; ++g)
#pragma unroll
    for (int j = 0; j < 8; ++j) a[g * 8 + j] = (bf16_t)p[g * 16 + j];
  return a;
}

__device__ __forceinline__ v16bf frag_a_from_bf16(const bf16_t* tile, int ld) {
  const int lane = threadIdx.x & 31;
  const int m = lane & 15, hf = lane >> 4;
  const bf16_t* p = tile + (size_t)m * ld + hf * 8;
  v16bf a;
#pragma unroll
  for (int g = 0; g < 2; ++g)
#pragma unroll
    for (int j = 0; j < 8; ++j) a[g * 8 + j] = p[g * 16 + j];
  return a;
}

// B (32x16, KxN): lane = col (n = lane&15), k = g*16 + half*8 + j.
// Column n of the tile lives at tile + n*ld (K-contiguous).
__device__ __forceinline__ v16bf frag_b_from_bf16(const bf16_t* tile, int ld) {
  const int lane = threadIdx.x & 31;
  const int n = lane & 15, hf = lane >> 4;
  const bf16_t* p = tile + (size_t)n * ld + hf * 8;
  v16bf b;
#pragma unroll
  for (int g = 0; g < 2; ++g)
#pragma unroll
    for (int j = 0; j < 8; ++j) b[g * 8 + j] = p[g * 16 + j];
  return b;
}

// ---------------------------------------------------------------------------
// Kernel 1: QKV projections.  Y = X @ W^T + bias  (fp32 in, bf16 out)
// grid = (D/64, (B*S)/128, 3), block = 256 (8 waves, each a 16x64 strip).
// Double-buffered LDS W-tile; per-thread stage = 8 contiguous elements so the
// LDS write is a single ds_store_b128 (16B aligned).  One barrier per k-step.
// z==0 -> q head-major [B,H,S,HD]; z==1 -> k head-major; z==2 -> v transposed
// [B,H,HD,S] so the ctx GEMM's B-operand is K-contiguous.
// ---------------------------------------------------------------------------
__global__ __launch_bounds__(256) void qkv_proj_kernel(
    const float* __restrict__ query, const float* __restrict__ key,
    const float* __restrict__ value, const float* __restrict__ Wq,
    const float* __restrict__ bq, const float* __restrict__ Wk,
    const float* __restrict__ bk, const float* __restrict__ Wv,
    const float* __restrict__ bv, bf16_t* __restrict__ qh,
    bf16_t* __restrict__ kh, bf16_t* __restrict__ vT) {
  const int which = blockIdx.z;
  const float *X, *W, *bias;
  if (which == 0) { X = query; W = Wq; bias = bq; }
  else if (which == 1) { X = key; W = Wk; bias = bk; }
  else { X = value; W = Wv; bias = bv; }

  const int wave = threadIdx.x >> 5;
  const int lane = threadIdx.x & 31;
  const int hf = lane >> 4, n16 = lane & 15;
  const int rowbase = blockIdx.y * 128 + wave * 16;
  const int colbase = blockIdx.x * 64;

  __shared__ bf16_t lb[2][64 * 32];  // double-buffered W tile [n=64][k=32]

  // stage mapping: thread owns tile elements [tid*8 .. tid*8+7] (contiguous)
  const int snn = threadIdx.x >> 2;        // tile row 0..63
  const int skk = (threadIdx.x & 3) * 8;   // tile col base 0/8/16/24
  const float* wrow = W + (size_t)(colbase + snn) * D_ + skk;
  float wreg[8];

  // prologue: stage k-tile 0 into buffer 0
#pragma unroll
  for (int r = 0; r < 8; ++r) wreg[r] = wrow[r];
  {
    v8bf tmp;
#pragma unroll
    for (int r = 0; r < 8; ++r) tmp[r] = (bf16_t)wreg[r];
    *reinterpret_cast<v8bf*>(&lb[0][threadIdx.x * 8]) = tmp;
  }
  __syncthreads();

  v8f acc[4] = {};

  for (int kb = 0; kb < D_; kb += 32) {
    const int cur = (kb >> 5) & 1;
    const bool have_next = (kb + 32) < D_;

    // issue next tile's global loads first (overlap with WMMAs below)
    if (have_next) {
#pragma unroll
      for (int r = 0; r < 8; ++r) wreg[r] = wrow[kb + 32 + r];
    }

    v16bf a  = frag_a_from_f32(X + (size_t)rowbase * D_ + kb, D_);
    v16bf b0 = frag_b_from_bf16(&lb[cur][0 * 512], 32);
    v16bf b1 = frag_b_from_bf16(&lb[cur][1 * 512], 32);
    v16bf b2 = frag_b_from_bf16(&lb[cur][2 * 512], 32);
    v16bf b3 = frag_b_from_bf16(&lb[cur][3 * 512], 32);
    acc[0] = WMMA_BF16(a, b0, acc[0]);
    acc[1] = WMMA_BF16(a, b1, acc[1]);
    acc[2] = WMMA_BF16(a, b2, acc[2]);
    acc[3] = WMMA_BF16(a, b3, acc[3]);

    if (have_next) {
      v8bf tmp;
#pragma unroll
      for (int r = 0; r < 8; ++r) tmp[r] = (bf16_t)wreg[r];
      *reinterpret_cast<v8bf*>(&lb[cur ^ 1][threadIdx.x * 8]) = tmp;
    }
    __syncthreads();
  }

#pragma unroll
  for (int t = 0; t < 4; ++t) {
    const int col = colbase + t * 16 + n16;
    const float bs = bias[col];
    const int h = col >> 6, hd = col & 63;
#pragma unroll
    for (int r = 0; r < 8; ++r) {
      const int m = rowbase + r + 8 * hf;   // global row in [0, B*S)
      const int b = m >> 11, s = m & (S_ - 1);
      const bf16_t val = (bf16_t)(acc[t][r] + bs);
      if (which == 2) {
        vT[((size_t)(b * H_ + h) * HD_ + hd) * S_ + s] = val;
      } else {
        bf16_t* dst = (which == 0) ? qh : kh;
        dst[((size_t)(b * H_ + h) * S_ + s) * HD_ + hd] = val;
      }
    }
  }
}

// ---------------------------------------------------------------------------
// Kernel 2: fused scores + softmax.  One WG per (b*h, 16-row stripe).
// 8 waves; wave w owns col-tiles {w, w+8, ..., w+120} -> 16 accum tiles held
// entirely in VGPRs.  Row reductions: 16-lane shfl_xor butterfly (rows 0-7 in
// lanes 0-15, rows 8-15 in lanes 16-31 per the C-layout) + LDS cross-wave
// reduce.  Writes the 537MB attention tensor exactly once.
// grid = (S/16, B*H), block = 256.
// ---------------------------------------------------------------------------
__global__ __launch_bounds__(256) void attn_softmax_kernel(
    const bf16_t* __restrict__ qh, const bf16_t* __restrict__ kh,
    float* __restrict__ attn) {
  const int bh = blockIdx.y;
  const int rowtile = blockIdx.x;
  const int wave = threadIdx.x >> 5;
  const int lane = threadIdx.x & 31;
  const int hf = lane >> 4, n16 = lane & 15;

  const bf16_t* qbase = qh + (size_t)bh * S_ * HD_ + (size_t)rowtile * 16 * HD_;
  const bf16_t* kbase = kh + (size_t)bh * S_ * HD_;
  float* arow = attn + (size_t)bh * S_ * S_ + (size_t)rowtile * 16 * S_;

  // Q stripe A-fragments (K = HD = 64 -> two k-steps), reused by all tiles
  const v16bf a0 = frag_a_from_bf16(qbase + 0, HD_);
  const v16bf a1 = frag_a_from_bf16(qbase + 32, HD_);

  v8f acc[16];
#pragma unroll
  for (int t = 0; t < 16; ++t) {
    const int ct = wave + t * 8;  // column tile 0..127
    const bf16_t* cb = kbase + (size_t)ct * 16 * HD_;
    // prefetch next K column-tile (2KB spread across 32 lanes, 64B each)
    if (t + 1 < 16)
      __builtin_prefetch(cb + (size_t)8 * 16 * HD_ + (size_t)lane * 32, 0, 0);
    v16bf b0 = frag_b_from_bf16(cb + 0, HD_);
    v16bf b1 = frag_b_from_bf16(cb + 32, HD_);
    v8f c = {};
    c = WMMA_BF16(a0, b0, c);
    c = WMMA_BF16(a1, b1, c);
#pragma unroll
    for (int r = 0; r < 8; ++r) c[r] *= INV_SCALE;
    acc[t] = c;
  }

  __shared__ float sred[8][16];

  // ---- row max ----
  float pm[8];
#pragma unroll
  for (int r = 0; r < 8; ++r) pm[r] = acc[0][r];
#pragma unroll
  for (int t = 1; t < 16; ++t)
#pragma unroll
    for (int r = 0; r < 8; ++r) pm[r] = fmaxf(pm[r], acc[t][r]);
#pragma unroll
  for (int r = 0; r < 8; ++r)
#pragma unroll
    for (int msk = 1; msk < 16; msk <<= 1)
      pm[r] = fmaxf(pm[r], __shfl_xor(pm[r], msk, 32));
  if (n16 == 0) {
#pragma unroll
    for (int r = 0; r < 8; ++r) sred[wave][hf * 8 + r] = pm[r];
  }
  __syncthreads();
  float fmx[8];
#pragma unroll
  for (int r = 0; r < 8; ++r) {
    float v = sred[0][hf * 8 + r];
#pragma unroll
    for (int w = 1; w < 8; ++w) v = fmaxf(v, sred[w][hf * 8 + r]);
    fmx[r] = v;
  }
  __syncthreads();

  // ---- exp + row sum ----
  float ps[8] = {};
#pragma unroll
  for (int t = 0; t < 16; ++t)
#pragma unroll
    for (int r = 0; r < 8; ++r) {
      float e = __expf(acc[t][r] - fmx[r]);
      acc[t][r] = e;
      ps[r] += e;
    }
#pragma unroll
  for (int r = 0; r < 8; ++r)
#pragma unroll
    for (int msk = 1; msk < 16; msk <<= 1) ps[r] += __shfl_xor(ps[r], msk, 32);
  if (n16 == 0) {
#pragma unroll
    for (int r = 0; r < 8; ++r) sred[wave][hf * 8 + r] = ps[r];
  }
  __syncthreads();
  float inv[8];
#pragma unroll
  for (int r = 0; r < 8; ++r) {
    float v = 0.f;
#pragma unroll
    for (int w = 0; w < 8; ++w) v += sred[w][hf * 8 + r];
    inv[r] = 1.0f / v;
  }

  // ---- normalize + single write of the attention tensor ----
#pragma unroll
  for (int t = 0; t < 16; ++t) {
    const int ct = wave + t * 8;
#pragma unroll
    for (int r = 0; r < 8; ++r)
      arow[(size_t)(r + 8 * hf) * S_ + ct * 16 + n16] = acc[t][r] * inv[r];
  }
}

// ---------------------------------------------------------------------------
// Kernel 3: ctx = P @ v  per (b,h).  P fp32 (read back from d_out, converted
// to bf16 in-register), v transposed bf16 [HD][S] staged through a
// double-buffered LDS tile (one global_load_b128 + one ds_store_b128 per
// thread per k-step).  grid = (S/128, B*H), block = 256; wave = 16 rows x
// 64 cols (full head dim).  Output: ctx bf16 [B,S,D].
// ---------------------------------------------------------------------------
__global__ __launch_bounds__(256) void ctx_kernel(
    const float* __restrict__ attn, const bf16_t* __restrict__ vT,
    bf16_t* __restrict__ ctx) {
  const int bh = blockIdx.y;
  const int wave = threadIdx.x >> 5;
  const int lane = threadIdx.x & 31;
  const int hf = lane >> 4, n16 = lane & 15;
  const int rowbase = blockIdx.x * 128 + wave * 16;  // query row within S

  const float* pbase = attn + (size_t)bh * S_ * S_;
  const bf16_t* vbase = vT + (size_t)bh * HD_ * S_;

  __shared__ bf16_t lb[2][64 * 32];  // double-buffered vT tile [hd=64][k=32]

  const int snn = threadIdx.x >> 2;
  const int skk = (threadIdx.x & 3) * 8;
  const bf16_t* vrow = vbase + (size_t)snn * S_ + skk;
  v8bf vreg;

  // prologue: stage k-tile 0 into buffer 0 (16B global load, 16B ds store)
  vreg = *reinterpret_cast<const v8bf*>(vrow);
  *reinterpret_cast<v8bf*>(&lb[0][threadIdx.x * 8]) = vreg;
  __syncthreads();

  v8f acc[4] = {};

  for (int kb = 0; kb < S_; kb += 32) {
    const int cur = (kb >> 5) & 1;
    const bool have_next = (kb + 32) < S_;

    if (have_next) vreg = *reinterpret_cast<const v8bf*>(vrow + kb + 32);

    v16bf a  = frag_a_from_f32(pbase + (size_t)rowbase * S_ + kb, S_);
    v16bf b0 = frag_b_from_bf16(&lb[cur][0 * 512], 32);
    v16bf b1 = frag_b_from_bf16(&lb[cur][1 * 512], 32);
    v16bf b2 = frag_b_from_bf16(&lb[cur][2 * 512], 32);
    v16bf b3 = frag_b_from_bf16(&lb[cur][3 * 512], 32);
    acc[0] = WMMA_BF16(a, b0, acc[0]);
    acc[1] = WMMA_BF16(a, b1, acc[1]);
    acc[2] = WMMA_BF16(a, b2, acc[2]);
    acc[3] = WMMA_BF16(a, b3, acc[3]);

    if (have_next)
      *reinterpret_cast<v8bf*>(&lb[cur ^ 1][threadIdx.x * 8]) = vreg;
    __syncthreads();
  }

  const int b = bh >> 4, h = bh & 15;
#pragma unroll
  for (int t = 0; t < 4; ++t)
#pragma unroll
    for (int r = 0; r < 8; ++r) {
      const int s = rowbase + r + 8 * hf;
      ctx[(size_t)(b * S_ + s) * D_ + h * HD_ + t * 16 + n16] =
          (bf16_t)acc[t][r];
    }
}

// ---------------------------------------------------------------------------
// Kernel 4: out = ctx @ Wo^T + bo  (bf16 x bf16 -> fp32 out), double-buffered
// LDS W-tile with b128 staging.  grid = (D/64, (B*S)/128), block = 256.
// ---------------------------------------------------------------------------
__global__ __launch_bounds__(256) void out_proj_kernel(
    const bf16_t* __restrict__ ctx, const float* __restrict__ Wo,
    const float* __restrict__ bo, float* __restrict__ out) {
  const int wave = threadIdx.x >> 5;
  const int lane = threadIdx.x & 31;
  const int hf = lane >> 4, n16 = lane & 15;
  const int rowbase = blockIdx.y * 128 + wave * 16;
  const int colbase = blockIdx.x * 64;

  __shared__ bf16_t lb[2][64 * 32];

  const int snn = threadIdx.x >> 2;
  const int skk = (threadIdx.x & 3) * 8;
  const float* wrow = Wo + (size_t)(colbase + snn) * D_ + skk;
  float wreg[8];

#pragma unroll
  for (int r = 0; r < 8; ++r) wreg[r] = wrow[r];
  {
    v8bf tmp;
#pragma unroll
    for (int r = 0; r < 8; ++r) tmp[r] = (bf16_t)wreg[r];
    *reinterpret_cast<v8bf*>(&lb[0][threadIdx.x * 8]) = tmp;
  }
  __syncthreads();

  v8f acc[4] = {};

  for (int kb = 0; kb < D_; kb += 32) {
    const int cur = (kb >> 5) & 1;
    const bool have_next = (kb + 32) < D_;

    if (have_next) {
#pragma unroll
      for (int r = 0; r < 8; ++r) wreg[r] = wrow[kb + 32 + r];
    }

    v16bf a  = frag_a_from_bf16(ctx + (size_t)rowbase * D_ + kb, D_);
    v16bf b0 = frag_b_from_bf16(&lb[cur][0 * 512], 32);
    v16bf b1 = frag_b_from_bf16(&lb[cur][1 * 512], 32);
    v16bf b2 = frag_b_from_bf16(&lb[cur][2 * 512], 32);
    v16bf b3 = frag_b_from_bf16(&lb[cur][3 * 512], 32);
    acc[0] = WMMA_BF16(a, b0, acc[0]);
    acc[1] = WMMA_BF16(a, b1, acc[1]);
    acc[2] = WMMA_BF16(a, b2, acc[2]);
    acc[3] = WMMA_BF16(a, b3, acc[3]);

    if (have_next) {
      v8bf tmp;
#pragma unroll
      for (int r = 0; r < 8; ++r) tmp[r] = (bf16_t)wreg[r];
      *reinterpret_cast<v8bf*>(&lb[cur ^ 1][threadIdx.x * 8]) = tmp;
    }
    __syncthreads();
  }

#pragma unroll
  for (int t = 0; t < 4; ++t) {
    const int col = colbase + t * 16 + n16;
    const float bs = bo[col];
#pragma unroll
    for (int r = 0; r < 8; ++r)
      out[(size_t)(rowbase + r + 8 * hf) * D_ + col] = acc[t][r] + bs;
  }
}

// ---------------------------------------------------------------------------
extern "C" void kernel_launch(void* const* d_in, const int* in_sizes, int n_in,
                              void* d_out, int out_size, void* d_ws,
                              size_t ws_size, hipStream_t stream) {
  (void)in_sizes; (void)n_in; (void)out_size; (void)ws_size;

  const float* query = (const float*)d_in[0];
  const float* key   = (const float*)d_in[1];
  const float* value = (const float*)d_in[2];
  const float* Wq    = (const float*)d_in[3];
  const float* bq    = (const float*)d_in[4];
  const float* Wk    = (const float*)d_in[5];
  const float* bk    = (const float*)d_in[6];
  const float* Wv    = (const float*)d_in[7];
  const float* bv    = (const float*)d_in[8];
  const float* Wo    = (const float*)d_in[9];
  const float* bo    = (const float*)d_in[10];

  float* out  = (float*)d_out;
  float* attn = out + (size_t)B_ * S_ * D_;  // second return value, flat

  // workspace: bf16 q, k, vT, ctx (8 MiB each = 32 MiB)
  bf16_t* qh  = (bf16_t*)d_ws;
  bf16_t* kh  = qh + (size_t)B_ * H_ * S_ * HD_;
  bf16_t* vT  = kh + (size_t)B_ * H_ * S_ * HD_;
  bf16_t* ctx = vT + (size_t)B_ * H_ * S_ * HD_;

  qkv_proj_kernel<<<dim3(D_ / 64, (B_ * S_) / 128, 3), 256, 0, stream>>>(
      query, key, value, Wq, bq, Wk, bk, Wv, bv, qh, kh, vT);

  attn_softmax_kernel<<<dim3(S_ / 16, B_ * H_), 256, 0, stream>>>(qh, kh, attn);

  ctx_kernel<<<dim3(S_ / 128, B_ * H_), 256, 0, stream>>>(attn, vT, ctx);

  out_proj_kernel<<<dim3(D_ / 64, (B_ * S_) / 128), 256, 0, stream>>>(
      ctx, Wo, bo, out);
}
